// BigBirdEncoderBlock_44203803410664
// MI455X (gfx1250) — matrix operand order repaired and checked by reading, not compiled, not run on hardware
//
#include <hip/hip_runtime.h>

// ---------------------------------------------------------------------------
// BigBird encoder block for MI455X (gfx1250, wave32, WMMA bf16 path)
// ---------------------------------------------------------------------------
#define B_   8
#define L_   1024
#define D_   512
#define H_   8
#define HD_  64
#define BLK_ 64
#define M_   16
#define S_   8
#define MLP_ 1024
#define NEGINF (-1e9f)

typedef __bf16 bf16_t;
typedef bf16_t v16bf __attribute__((ext_vector_type(16)));
typedef float  v8f   __attribute__((ext_vector_type(8)));

union FragB16 { v16bf v; unsigned int u[8]; };
union Vec8    { v8f   v; float        f[8]; };

static __device__ __forceinline__ unsigned short f2bf(float f) {
  unsigned int x = __float_as_uint(f);
  x += 0x7fffu + ((x >> 16) & 1u);            // round-to-nearest-even
  return (unsigned short)(x >> 16);
}

// K offset of bf16 pair i (VGPR i) for the A-matrix 16x32 layout (ISA 7.12.2):
// lanes 0-15: VGPR0-3 -> K 0..7, VGPR4-7 -> K 16..23 ; lanes 16-31: +8
static __device__ __forceinline__ int a_pair_k(int i, int half) {
  return ((i & 3) << 1) + ((i >> 2) << 4) + (half << 3);
}

static __device__ __forceinline__ v8f wmma_bf16(v16bf a, v16bf b, v8f c) {
  return __builtin_amdgcn_wmma_f32_16x16x32_bf16(false, a, false, b,
                                                 (short)0, c, false, false);
}

static __device__ __forceinline__ float gelu_tanh(float x) {
  float x3 = x * x * x;
  return 0.5f * x * (1.f + tanhf(0.7978845608028654f * (x + 0.044715f * x3)));
}

// ---------------------------------------------------------------------------
// fp32 [K][N] -> bf16 [N][K] transposed weight conversion (LDS-tiled)
// ---------------------------------------------------------------------------
__global__ void __launch_bounds__(256)
cvt_bf16_T_kernel(const float* __restrict__ in, unsigned short* __restrict__ out,
                  int K, int N) {
  __shared__ float tile[32][33];
  int kt = blockIdx.x * 32, nt = blockIdx.y * 32;
  int tx = threadIdx.x & 31, ty = threadIdx.x >> 5;   // 32 x 8
#pragma unroll
  for (int i = 0; i < 32; i += 8)
    tile[ty + i][tx] = in[(size_t)(kt + ty + i) * N + nt + tx];
  __syncthreads();
#pragma unroll
  for (int i = 0; i < 32; i += 8)
    out[(size_t)(nt + ty + i) * K + kt + tx] = f2bf(tile[tx][ty + i]);
}

// ---------------------------------------------------------------------------
// BigBird gathered-block index / dedup-keep table
// ---------------------------------------------------------------------------
__global__ void setup_idx_kernel(const int* __restrict__ rand_attn,
                                 int* __restrict__ idx, int* __restrict__ keep) {
  int i = threadIdx.x;
  if (i >= M_) return;
  int t[S_];
  t[0] = 0; t[1] = M_ - 1; t[2] = (i + M_ - 1) % M_; t[3] = i; t[4] = (i + 1) % M_;
  t[5] = rand_attn[i * 3 + 0];
  t[6] = rand_attn[i * 3 + 1];
  t[7] = rand_attn[i * 3 + 2];
  for (int s = 0; s < S_; ++s) {
    int k = 1;
    for (int ss = 0; ss < s; ++ss) if (t[ss] == t[s]) k = 0;
    idx[i * S_ + s] = t[s];
    keep[i * S_ + s] = k;
  }
}

// ---------------------------------------------------------------------------
// LayerNorm (one wave per row of 512) -> bf16 output
// ---------------------------------------------------------------------------
__global__ void __launch_bounds__(256)
ln_bf16_kernel(const float* __restrict__ x, const float* __restrict__ sc,
               const float* __restrict__ bi, unsigned short* __restrict__ out,
               int rows) {
  int wid = blockIdx.x * 8 + (threadIdx.x >> 5);
  if (wid >= rows) return;
  int lane = threadIdx.x & 31;
  const float* rp = x + (size_t)wid * D_;

  float4 v[4];
  float s = 0.f;
#pragma unroll
  for (int c = 0; c < 4; ++c) {
    v[c] = *(const float4*)(rp + 4 * lane + 128 * c);
    s += v[c].x + v[c].y + v[c].z + v[c].w;
  }
#pragma unroll
  for (int off = 16; off >= 1; off >>= 1) s += __shfl_xor(s, off, 32);
  float mean = s * (1.f / D_);

  float t = 0.f;
#pragma unroll
  for (int c = 0; c < 4; ++c) {
    float dx = v[c].x - mean, dy = v[c].y - mean, dz = v[c].z - mean, dw = v[c].w - mean;
    t += dx * dx + dy * dy + dz * dz + dw * dw;
  }
#pragma unroll
  for (int off = 16; off >= 1; off >>= 1) t += __shfl_xor(t, off, 32);
  float rstd = rsqrtf(t * (1.f / D_) + 1e-6f);

  unsigned short* op = out + (size_t)wid * D_;
#pragma unroll
  for (int c = 0; c < 4; ++c) {
    int col = 4 * lane + 128 * c;
    op[col + 0] = f2bf((v[c].x - mean) * rstd * sc[col + 0] + bi[col + 0]);
    op[col + 1] = f2bf((v[c].y - mean) * rstd * sc[col + 1] + bi[col + 1]);
    op[col + 2] = f2bf((v[c].z - mean) * rstd * sc[col + 2] + bi[col + 2]);
    op[col + 3] = f2bf((v[c].w - mean) * rstd * sc[col + 3] + bi[col + 3]);
  }
}

// ---------------------------------------------------------------------------
// bf16 GEMM: C[M,N] = A[M,K] * Bt[N,K]^T. 128x64 block tile, 4 waves, each
// wave owns a 32x64 strip (2 A-frags x 4 B-frags = 8 WMMA / K-step).
// Double-buffered LDS pipeline; staging lives in named uint4 VGPRs (macros,
// no lambdas / no local arrays -> no scratch spills).
// Epilogues:
//   1: bf16, scale, transposed store to [B,H,L,HD]   (Q / K projections)
//   5: bf16, transposed store to [B,H,HD,L]          (V projection)
//   2: fp32, + residual                              (Wo projection -> xr)
//   3: bf16, + bias, GELU(tanh)                      (MLP1)
//   4: fp32, + bias, + residual                      (MLP2 -> final output)
// ---------------------------------------------------------------------------
#define G_LOAD(k0)                                                              \
  do {                                                                          \
    const unsigned short* ap_ = A + (size_t)(m0 + (tid >> 2)) * K + (k0) + (tid & 3) * 8; \
    ra0 = *(const uint4*)(ap_);                                                 \
    ra1 = *(const uint4*)(ap_ + (size_t)32 * K);                                \
    ra2 = *(const uint4*)(ap_ + (size_t)64 * K);                                \
    ra3 = *(const uint4*)(ap_ + (size_t)96 * K);                                \
    const unsigned short* bp_ = Bt + (size_t)(n0 + (tid >> 2)) * K + (k0) + (tid & 3) * 8; \
    rb0 = *(const uint4*)(bp_);                                                 \
    rb1 = *(const uint4*)(bp_ + (size_t)32 * K);                                \
  } while (0)

#define S_STORE(buf)                                                            \
  do {                                                                          \
    const int row_ = tid >> 2, kc_ = tid & 3;                                   \
    ((uint4*)(&As[buf][row_ +  0][0]))[kc_] = ra0;                              \
    ((uint4*)(&As[buf][row_ + 32][0]))[kc_] = ra1;                              \
    ((uint4*)(&As[buf][row_ + 64][0]))[kc_] = ra2;                              \
    ((uint4*)(&As[buf][row_ + 96][0]))[kc_] = ra3;                              \
    ((uint4*)(&Bs[buf][row_ +  0][0]))[kc_] = rb0;                              \
    ((uint4*)(&Bs[buf][row_ + 32][0]))[kc_] = rb1;                              \
  } while (0)

__global__ void __launch_bounds__(128)
gemm_bf16_kernel(const unsigned short* __restrict__ A,   // [M][K] bf16
                 const unsigned short* __restrict__ Bt,  // [N][K] bf16 (pre-transposed)
                 int Mrows, int N, int K,
                 const float* __restrict__ bias,
                 const float* __restrict__ resid,
                 float* __restrict__ outf,
                 unsigned short* __restrict__ outb,
                 int mode, float scale) {
  __shared__ unsigned short As[2][128][40];   // [m][k], 80B row stride
  __shared__ unsigned short Bs[2][64][40];    // [n][k]

  const int m0 = blockIdx.x * 128;
  const int n0 = blockIdx.y * 64;
  const int tid = threadIdx.x;
  const int w = tid >> 5, lane = tid & 31, ln = lane & 15, half = lane >> 4;

  uint4 ra0, ra1, ra2, ra3, rb0, rb1;

  Vec8 acc[2][4];
#pragma unroll
  for (int rt = 0; rt < 2; ++rt)
#pragma unroll
    for (int nt = 0; nt < 4; ++nt)
#pragma unroll
      for (int r = 0; r < 8; ++r) acc[rt][nt].f[r] = 0.f;

  G_LOAD(0);
  S_STORE(0);
  const int nk = K >> 5;
  for (int ki = 0; ki < nk; ++ki) {
    __syncthreads();
    const bool more = (ki + 1 < nk);
    if (more) G_LOAD((ki + 1) << 5);          // overlap with WMMA below

    const int cur = ki & 1;
    FragB16 af[2];
#pragma unroll
    for (int rt = 0; rt < 2; ++rt)
#pragma unroll
      for (int i = 0; i < 8; ++i)
        af[rt].u[i] =
            *(const unsigned int*)(&As[cur][w * 32 + rt * 16 + ln][a_pair_k(i, half)]);
#pragma unroll
    for (int nt = 0; nt < 4; ++nt) {
      FragB16 bf;
#pragma unroll
      for (int i = 0; i < 8; ++i)
        bf.u[i] = *(const unsigned int*)(&Bs[cur][nt * 16 + ln][a_pair_k(i, half)]);
      acc[0][nt].v = wmma_bf16(af[0].v, bf.v, acc[0][nt].v);
      acc[1][nt].v = wmma_bf16(af[1].v, bf.v, acc[1][nt].v);
    }
    if (more) {
      __syncthreads();
      S_STORE(cur ^ 1);
    }
  }

  // C layout: VGPR r, lanes 0-15 -> row r, lanes 16-31 -> row r+8
#pragma unroll
  for (int rt = 0; rt < 2; ++rt)
#pragma unroll
    for (int nt = 0; nt < 4; ++nt)
#pragma unroll
      for (int r = 0; r < 8; ++r) {
        int grow = m0 + w * 32 + rt * 16 + r + (half << 3);
        int gcol = n0 + nt * 16 + ln;
        float v = acc[rt][nt].f[r];
        if (mode == 1) {          // Q/K: [B*L, H*HD] -> bf16 [B,H,L,HD], scaled
          int b = grow >> 10, l = grow & 1023, hh = gcol >> 6, hd = gcol & 63;
          outb[(((size_t)(b * H_ + hh) * L_ + l) << 6) + hd] = f2bf(v * scale);
        } else if (mode == 5) {   // V: -> bf16 [B,H,HD,L] (transposed for P*V)
          int b = grow >> 10, l = grow & 1023, hh = gcol >> 6, hd = gcol & 63;
          outb[((size_t)(b * H_ + hh) * HD_ + hd) * L_ + l] = f2bf(v);
        } else if (mode == 2) {   // Wo: fp32 + residual
          size_t o = (size_t)grow * N + gcol;
          outf[o] = v + resid[o];
        } else if (mode == 3) {   // MLP1: bias + GELU -> bf16
          outb[(size_t)grow * N + gcol] = f2bf(gelu_tanh(v + bias[gcol]));
        } else {                  // MLP2: bias + residual -> fp32
          size_t o = (size_t)grow * N + gcol;
          outf[o] = v + bias[gcol] + resid[o];
        }
      }
}

// ---------------------------------------------------------------------------
// BigBird sparse attention: one workgroup per (b,h,m) query block (64 rows).
// 4 waves x 16 query rows. Flash-style online softmax over gathered 64-key
// blocks (or all 16 blocks for the global rows m==0 / m==M-1).
// K/V block loads are double-buffered through LDS; valid-block list lives in
// LDS (uniform); staging in named uint4 VGPRs -> no scratch.
// ---------------------------------------------------------------------------
#define ATT_LOAD(kblk)                                                          \
  do {                                                                          \
    const unsigned short* kp_ =                                                 \
        kb + (bh + (size_t)(kblk) * 64 + (tid >> 3)) * HD_ + (tid & 7) * 8;     \
    rk0 = *(const uint4*)(kp_);                                                 \
    rk1 = *(const uint4*)(kp_ + 16 * HD_);                                      \
    rk2 = *(const uint4*)(kp_ + 32 * HD_);                                      \
    rk3 = *(const uint4*)(kp_ + 48 * HD_);                                      \
    const unsigned short* vp_ =                                                 \
        vbase + (size_t)(tid >> 3) * L_ + (size_t)(kblk) * 64 + (tid & 7) * 8;  \
    rv0 = *(const uint4*)(vp_);                                                 \
    rv1 = *(const uint4*)(vp_ + 16 * L_);                                       \
    rv2 = *(const uint4*)(vp_ + 32 * L_);                                       \
    rv3 = *(const uint4*)(vp_ + 48 * L_);                                       \
  } while (0)

#define ATT_STORE(buf)                                                          \
  do {                                                                          \
    const int row_ = tid >> 3, kc_ = tid & 7;                                   \
    ((uint4*)(&Ks[buf][row_ +  0][0]))[kc_] = rk0;                              \
    ((uint4*)(&Ks[buf][row_ + 16][0]))[kc_] = rk1;                              \
    ((uint4*)(&Ks[buf][row_ + 32][0]))[kc_] = rk2;                              \
    ((uint4*)(&Ks[buf][row_ + 48][0]))[kc_] = rk3;                              \
    ((uint4*)(&Vs[buf][row_ +  0][0]))[kc_] = rv0;                              \
    ((uint4*)(&Vs[buf][row_ + 16][0]))[kc_] = rv1;                              \
    ((uint4*)(&Vs[buf][row_ + 32][0]))[kc_] = rv2;                              \
    ((uint4*)(&Vs[buf][row_ + 48][0]))[kc_] = rv3;                              \
  } while (0)

__global__ void __launch_bounds__(128)
bigbird_attn_kernel(const unsigned short* __restrict__ qb,   // [B,H,L,HD] bf16 (pre-scaled)
                    const unsigned short* __restrict__ kb,   // [B,H,L,HD] bf16
                    const unsigned short* __restrict__ vt,   // [B,H,HD,L] bf16
                    const unsigned char* __restrict__ pmask, // [B,L] bool
                    const int* __restrict__ idx,             // [M,S]
                    const int* __restrict__ keep,            // [M,S]
                    unsigned short* __restrict__ attnb)      // [B,L,H*HD] bf16
{
  __shared__ unsigned short Ks[2][64][72];    // [key][hd]  == B^T for Q*K^T
  __shared__ unsigned short Vs[2][64][72];    // [hd][key]  == B^T for P*V
  __shared__ unsigned short Ps[4][16][72];    // per-wave probs (A-layout source)
  __shared__ int sblist[M_ + 1];              // valid block list + count

  const int m = blockIdx.x, h = blockIdx.y, b = blockIdx.z;
  const int tid = threadIdx.x;
  const int w = tid >> 5, lane = tid & 31, ln = lane & 15, half = lane >> 4;
  const bool is_global = (m == 0) || (m == M_ - 1);
  const size_t bh = (size_t)(b * H_ + h) * L_;
  const unsigned short* vbase = vt + (size_t)(b * H_ + h) * HD_ * L_;

  if (tid == 0) {
    int nv = 0;
    if (is_global) {
      for (int j = 0; j < M_; ++j) sblist[nv++] = j;
    } else {
      for (int j = 0; j < S_; ++j)
        if (keep[m * S_ + j]) sblist[nv++] = idx[m * S_ + j];
    }
    sblist[M_] = nv;
  }

  // Q fragments (A layout), held in registers for the whole kernel
  FragB16 aq[2];
  {
    int qrow = m * 64 + w * 16 + ln;
    const unsigned short* qp = qb + (bh + qrow) * HD_;
#pragma unroll
    for (int ks = 0; ks < 2; ++ks)
#pragma unroll
      for (int i = 0; i < 8; ++i)
        aq[ks].u[i] = *(const unsigned int*)(qp + a_pair_k(i, half) + ks * 32);
  }

  float run_max[8], run_sum[8];
  Vec8 oacc[4];
#pragma unroll
  for (int r = 0; r < 8; ++r) { run_max[r] = -1e30f; run_sum[r] = 0.f; }
#pragma unroll
  for (int nt = 0; nt < 4; ++nt)
#pragma unroll
    for (int r = 0; r < 8; ++r) oacc[nt].f[r] = 0.f;

  __syncthreads();                     // sblist visible
  const int nv = sblist[M_];

  uint4 rk0, rk1, rk2, rk3, rv0, rv1, rv2, rv3;
  ATT_LOAD(sblist[0]);
  ATT_STORE(0);
  for (int jj = 0; jj < nv; ++jj) {
    __syncthreads();
    const bool more = (jj + 1 < nv);
    if (more) ATT_LOAD(sblist[jj + 1]);          // overlap with compute below
    const int cur = jj & 1;
    const int kblk = sblist[jj];

    // ---- scores = Q * K^T  (16 rows x 64 keys per wave)
    Vec8 sacc[4];
#pragma unroll
    for (int nt = 0; nt < 4; ++nt)
#pragma unroll
      for (int r = 0; r < 8; ++r) sacc[nt].f[r] = 0.f;
#pragma unroll
    for (int ks = 0; ks < 2; ++ks)
#pragma unroll
      for (int nt = 0; nt < 4; ++nt) {
        FragB16 bf;
#pragma unroll
        for (int i = 0; i < 8; ++i)
          bf.u[i] = *(const unsigned int*)(&Ks[cur][nt * 16 + ln]
                                              [a_pair_k(i, half) + (ks << 5)]);
        sacc[nt].v = wmma_bf16(aq[ks].v, bf.v, sacc[nt].v);
      }

    // ---- padding mask + per-row block max
    float bmax[8];
#pragma unroll
    for (int r = 0; r < 8; ++r) bmax[r] = -1e30f;
#pragma unroll
    for (int nt = 0; nt < 4; ++nt) {
      unsigned char p = pmask[(size_t)b * L_ + kblk * 64 + nt * 16 + ln];
#pragma unroll
      for (int r = 0; r < 8; ++r) {
        float s = p ? sacc[nt].f[r] : NEGINF;
        sacc[nt].f[r] = s;
        bmax[r] = fmaxf(bmax[r], s);
      }
    }
#pragma unroll
    for (int off = 1; off < 16; off <<= 1)
#pragma unroll
      for (int r = 0; r < 8; ++r) bmax[r] = fmaxf(bmax[r], __shfl_xor(bmax[r], off, 32));

    float corr[8], lsum[8];
#pragma unroll
    for (int r = 0; r < 8; ++r) {
      float nm = fmaxf(run_max[r], bmax[r]);
      corr[r] = __expf(run_max[r] - nm);
      run_max[r] = nm;
      lsum[r] = 0.f;
    }

    // ---- exponentiate, stash P (bf16) into LDS (C-layout -> A-layout bridge)
#pragma unroll
    for (int nt = 0; nt < 4; ++nt)
#pragma unroll
      for (int r = 0; r < 8; ++r) {
        float pv = __expf(sacc[nt].f[r] - run_max[r]);
        lsum[r] += pv;
        Ps[w][r + (half << 3)][nt * 16 + ln] = f2bf(pv);
      }
#pragma unroll
    for (int off = 1; off < 16; off <<= 1)
#pragma unroll
      for (int r = 0; r < 8; ++r) lsum[r] += __shfl_xor(lsum[r], off, 32);
#pragma unroll
    for (int r = 0; r < 8; ++r) run_sum[r] = run_sum[r] * corr[r] + lsum[r];
#pragma unroll
    for (int nt = 0; nt < 4; ++nt)
#pragma unroll
      for (int r = 0; r < 8; ++r) oacc[nt].f[r] *= corr[r];

    // ---- O += P * V  (contraction over 64 keys, 2 WMMA K-steps)
#pragma unroll
    for (int ks = 0; ks < 2; ++ks) {
      FragB16 ap;
#pragma unroll
      for (int i = 0; i < 8; ++i)
        ap.u[i] = *(const unsigned int*)(&Ps[w][ln][a_pair_k(i, half) + (ks << 5)]);
#pragma unroll
      for (int nt = 0; nt < 4; ++nt) {
        FragB16 bv;
#pragma unroll
        for (int i = 0; i < 8; ++i)
          bv.u[i] = *(const unsigned int*)(&Vs[cur][nt * 16 + ln]
                                              [a_pair_k(i, half) + (ks << 5)]);
        oacc[nt].v = wmma_bf16(ap.v, bv.v, oacc[nt].v);
      }
    }
    if (more) {
      __syncthreads();
      ATT_STORE(cur ^ 1);
    }
  }

  // ---- normalize + store bf16 to [B, L, H*HD]
#pragma unroll
  for (int r = 0; r < 8; ++r) run_sum[r] = (run_sum[r] > 0.f) ? 1.f / run_sum[r] : 0.f;
#pragma unroll
  for (int nt = 0; nt < 4; ++nt)
#pragma unroll
    for (int r = 0; r < 8; ++r) {
      int qrow = m * 64 + w * 16 + r + (half << 3);
      int col = h * HD_ + nt * 16 + ln;
      attnb[((size_t)b * L_ + qrow) * (H_ * HD_) + col] = f2bf(oacc[nt].f[r] * run_sum[r]);
    }
}

// ---------------------------------------------------------------------------
// host-side orchestration
// ---------------------------------------------------------------------------
extern "C" void kernel_launch(void* const* d_in, const int* in_sizes, int n_in,
                              void* d_out, int out_size, void* d_ws, size_t ws_size,
                              hipStream_t stream) {
  (void)in_sizes; (void)n_in; (void)out_size; (void)ws_size;

  const float* inputs = (const float*)d_in[0];
  const unsigned char* pmask = (const unsigned char*)d_in[1];
  const float* ln1s = (const float*)d_in[2];
  const float* ln1b = (const float*)d_in[3];
  const float* Wq = (const float*)d_in[4];
  const float* Wk = (const float*)d_in[5];
  const float* Wv = (const float*)d_in[6];
  const float* Wo = (const float*)d_in[7];
  const float* ln2s = (const float*)d_in[8];
  const float* ln2b = (const float*)d_in[9];
  const float* W1 = (const float*)d_in[10];
  const float* b1 = (const float*)d_in[11];
  const float* W2 = (const float*)d_in[12];
  const float* b2 = (const float*)d_in[13];
  const int* rand_attn = (const int*)d_in[14];
  float* out = (float*)d_out;

  const size_t ROWS = (size_t)B_ * L_;              // 8192
  char* ws = (char*)d_ws;
  size_t o = 0;
  auto alloc = [&](size_t bytes) {
    char* p = ws + o;
    o += (bytes + 255) & ~(size_t)255;
    return p;
  };
  unsigned short* xln  = (unsigned short*)alloc(ROWS * D_ * 2);
  unsigned short* qbf  = (unsigned short*)alloc((size_t)B_ * H_ * L_ * HD_ * 2);
  unsigned short* kbf  = (unsigned short*)alloc((size_t)B_ * H_ * L_ * HD_ * 2);
  unsigned short* vbt  = (unsigned short*)alloc((size_t)B_ * H_ * L_ * HD_ * 2);
  unsigned short* attn = (unsigned short*)alloc(ROWS * D_ * 2);
  float*          xr   = (float*)         alloc(ROWS * D_ * 4);
  unsigned short* yln  = (unsigned short*)alloc(ROWS * D_ * 2);
  unsigned short* hb   = (unsigned short*)alloc(ROWS * MLP_ * 2);
  unsigned short* WqT  = (unsigned short*)alloc((size_t)D_ * D_ * 2);
  unsigned short* WkT  = (unsigned short*)alloc((size_t)D_ * D_ * 2);
  unsigned short* WvT  = (unsigned short*)alloc((size_t)D_ * D_ * 2);
  unsigned short* WoT  = (unsigned short*)alloc((size_t)D_ * D_ * 2);
  unsigned short* W1T  = (unsigned short*)alloc((size_t)D_ * MLP_ * 2);
  unsigned short* W2T  = (unsigned short*)alloc((size_t)MLP_ * D_ * 2);
  int* idxd  = (int*)alloc(M_ * S_ * 4);
  int* keepd = (int*)alloc(M_ * S_ * 4);

  // weight conversion + transpose (fp32 [K][N] -> bf16 [N][K])
  cvt_bf16_T_kernel<<<dim3(D_ / 32, D_ / 32), 256, 0, stream>>>(Wq, WqT, D_, D_);
  cvt_bf16_T_kernel<<<dim3(D_ / 32, D_ / 32), 256, 0, stream>>>(Wk, WkT, D_, D_);
  cvt_bf16_T_kernel<<<dim3(D_ / 32, D_ / 32), 256, 0, stream>>>(Wv, WvT, D_, D_);
  cvt_bf16_T_kernel<<<dim3(D_ / 32, D_ / 32), 256, 0, stream>>>(Wo, WoT, D_, D_);
  cvt_bf16_T_kernel<<<dim3(D_ / 32, MLP_ / 32), 256, 0, stream>>>(W1, W1T, D_, MLP_);
  cvt_bf16_T_kernel<<<dim3(MLP_ / 32, D_ / 32), 256, 0, stream>>>(W2, W2T, MLP_, D_);
  setup_idx_kernel<<<1, 32, 0, stream>>>(rand_attn, idxd, keepd);

  // LN1 -> bf16
  ln_bf16_kernel<<<ROWS / 8, 256, 0, stream>>>(inputs, ln1s, ln1b, xln, (int)ROWS);

  // QKV projections (q pre-scaled by 1/sqrt(HD)); K -> [B,H,L,HD], V -> [B,H,HD,L]
  dim3 gqkv(ROWS / 128, D_ / 64);
  gemm_bf16_kernel<<<gqkv, 128, 0, stream>>>(xln, WqT, (int)ROWS, D_, D_,
                                             nullptr, nullptr, nullptr, qbf, 1, 0.125f);
  gemm_bf16_kernel<<<gqkv, 128, 0, stream>>>(xln, WkT, (int)ROWS, D_, D_,
                                             nullptr, nullptr, nullptr, kbf, 1, 1.0f);
  gemm_bf16_kernel<<<gqkv, 128, 0, stream>>>(xln, WvT, (int)ROWS, D_, D_,
                                             nullptr, nullptr, nullptr, vbt, 5, 1.0f);

  // sparse attention
  bigbird_attn_kernel<<<dim3(M_, H_, B_), 128, 0, stream>>>(qbf, kbf, vbt, pmask,
                                                            idxd, keepd, attn);

  // output projection + residual -> xr (fp32)
  gemm_bf16_kernel<<<dim3(ROWS / 128, D_ / 64), 128, 0, stream>>>(
      attn, WoT, (int)ROWS, D_, D_, nullptr, inputs, xr, nullptr, 2, 1.0f);

  // LN2 -> bf16
  ln_bf16_kernel<<<ROWS / 8, 256, 0, stream>>>(xr, ln2s, ln2b, yln, (int)ROWS);

  // MLP1: GELU(yln @ W1 + b1) -> bf16
  gemm_bf16_kernel<<<dim3(ROWS / 128, MLP_ / 64), 128, 0, stream>>>(
      yln, W1T, (int)ROWS, MLP_, D_, b1, nullptr, nullptr, hb, 3, 1.0f);

  // MLP2: h @ W2 + b2 + xr -> out (fp32)
  gemm_bf16_kernel<<<dim3(ROWS / 128, D_ / 64), 128, 0, stream>>>(
      hb, W2T, (int)ROWS, D_, MLP_, b2, xr, out, nullptr, 4, 1.0f);
}
